// Decoder_6992206758452
// MI455X (gfx1250) — matrix operand order repaired and checked by reading, not compiled, not run on hardware
//
#include <hip/hip_runtime.h>
#include <hip/hip_bf16.h>
#include <math.h>

// ---------------------------------------------------------------------------
// Types for WMMA fragments (gfx1250, wave32)
// ---------------------------------------------------------------------------
typedef _Float16 h16;
typedef __attribute__((ext_vector_type(8)))  _Float16 v8h;
typedef __attribute__((ext_vector_type(16))) _Float16 v16h;
typedef __attribute__((ext_vector_type(8)))  float    v8f;

// Problem constants
#define VV   32000
#define ED   512
#define HH   1024
#define BB   64
#define SRC_ 64
#define NSTEP 32          // TGT-1

// ---------------------------------------------------------------------------
// Fragment loaders.
// A-matrix 16x32 f16 (row m per lane): VGPR v0..3 hold K = h*8 + 2v(,+1),
// v4..7 hold K = 16 + h*8 + 2(v-4)(,+1), h = lane/16.
// B-matrix 32x16 f16 (col n per lane): same K layout as A, g = lane/16.
// Since B = W^T and W is (N x K) row-major, B column n is two 8-half
// chunks of W row n.
// ---------------------------------------------------------------------------
__device__ __forceinline__ v16h load_a_frag(const h16* __restrict__ rowA,
                                            int kk, int hsel) {
  v8h lo = *(const v8h*)(rowA + kk + hsel * 8);
  v8h hi = *(const v8h*)(rowA + kk + 16 + hsel * 8);
  v16h r;
#pragma unroll
  for (int i = 0; i < 8; ++i) { r[i] = lo[i]; r[i + 8] = hi[i]; }
  return r;
}

__device__ __forceinline__ v16h load_b_frag(const h16* __restrict__ rowW,
                                            int kk, int g) {
  v8h lo = *(const v8h*)(rowW + kk + g * 8);
  v8h hi = *(const v8h*)(rowW + kk + 16 + g * 8);
  v16h r;
#pragma unroll
  for (int i = 0; i < 8; ++i) { r[i] = lo[i]; r[i + 8] = hi[i]; }
  return r;
}

#define WMMA_F16(A, B, C) \
  __builtin_amdgcn_wmma_f32_16x16x32_f16(false, (A), false, (B), (short)0, (C), false, false)

// ---------------------------------------------------------------------------
// Generic GEMM:  C(MxN) = A(MxK,f16,rowmajor) * W(NxK,f16,rowmajor)^T [+bias]
// Block = 128 threads (4 waves); block tile 64x64; wave tile 32x32.
// M, N multiples of 64; K multiple of 32.
// epilogue: 0 = none, 1 = tanh
// ---------------------------------------------------------------------------
__global__ __launch_bounds__(128)
void gemm_f16_wmma(const h16* __restrict__ A, const h16* __restrict__ W,
                   const float* __restrict__ bias, float* __restrict__ C,
                   int M, int N, int K, int epilogue) {
  const int lane = threadIdx.x & 31;
  const int wave = threadIdx.x >> 5;
  const int lm   = lane & 15;
  const int g    = lane >> 4;

  const int m0 = blockIdx.y * 64 + (wave >> 1) * 32;
  const int n0 = blockIdx.x * 64 + (wave & 1) * 32;

  const h16* a0 = A + (size_t)(m0 + lm) * K;
  const h16* a1 = a0 + (size_t)16 * K;
  const h16* w0 = W + (size_t)(n0 + lm) * K;
  const h16* w1 = w0 + (size_t)16 * K;

  v8f acc00 = {}, acc01 = {}, acc10 = {}, acc11 = {};

  for (int kk = 0; kk < K; kk += 32) {
    v16h af0 = load_a_frag(a0, kk, g);
    v16h af1 = load_a_frag(a1, kk, g);
    v16h bf0 = load_b_frag(w0, kk, g);
    v16h bf1 = load_b_frag(w1, kk, g);
    acc00 = WMMA_F16(af0, bf0, acc00);
    acc01 = WMMA_F16(af0, bf1, acc01);
    acc10 = WMMA_F16(af1, bf0, acc10);
    acc11 = WMMA_F16(af1, bf1, acc11);
  }

  v8f* accs[2][2] = {{&acc00, &acc01}, {&acc10, &acc11}};
#pragma unroll
  for (int i = 0; i < 2; ++i) {
#pragma unroll
    for (int j = 0; j < 2; ++j) {
      const int col = n0 + 16 * j + lm;
      const float bv = bias ? bias[col] : 0.0f;
      float* cp = C + (size_t)(m0 + 16 * i + g * 8) * N + col;
      v8f a = *accs[i][j];
#pragma unroll
      for (int v = 0; v < 8; ++v) {
        float val = a[v] + bv;
        if (epilogue == 1) val = tanhf(val);
        cp[(size_t)v * N] = val;
      }
    }
  }
}

// ---------------------------------------------------------------------------
// Small elementwise / setup kernels
// ---------------------------------------------------------------------------
__global__ void f32_to_f16_k(const float* __restrict__ s, h16* __restrict__ d, int n) {
  int i = blockIdx.x * blockDim.x + threadIdx.x;
  if (i < n) d[i] = (h16)s[i];
}

// Wz (4096 x 2560) = [W_ih (4096x1536) | W_hh (4096x1024)] along K, in f16
__global__ void build_wz_k(const float* __restrict__ Wih, const float* __restrict__ Whh,
                           h16* __restrict__ Wz, int n) {
  int idx = blockIdx.x * blockDim.x + threadIdx.x;
  if (idx >= n) return;
  int r = idx / 2560, k = idx % 2560;
  float v = (k < 1536) ? Wih[(size_t)r * 1536 + k] : Whh[(size_t)r * 1024 + (k - 1536)];
  Wz[idx] = (h16)v;
}

__global__ void bias_sum_k(const float* __restrict__ a, const float* __restrict__ b,
                           float* __restrict__ o, int n) {
  int i = blockIdx.x * blockDim.x + threadIdx.x;
  if (i < n) o[i] = a[i] + b[i];
}

__global__ void init_state_k(const float* __restrict__ h0, const float* __restrict__ c0,
                             float* __restrict__ h, float* __restrict__ c,
                             float* __restrict__ o, int n) {
  int i = blockIdx.x * blockDim.x + threadIdx.x;
  if (i < n) { h[i] = h0[i]; c[i] = c0[i]; o[i] = 0.0f; }
}

// z (64 x 2560, f16) = [ E[tgt[t,b]] (512) | o_prev (1024) | h_prev (1024) ]
__global__ void build_z_k(const float* __restrict__ E, const int* __restrict__ tgt,
                          const float* __restrict__ o_prev, const float* __restrict__ h,
                          h16* __restrict__ z, int t) {
  int idx = blockIdx.x * blockDim.x + threadIdx.x;
  if (idx >= BB * 2560) return;
  int b = idx / 2560, k = idx % 2560;
  float v;
  if (k < ED) {
    int tok = tgt[t * BB + b];
    v = E[(size_t)tok * ED + k];
  } else if (k < ED + HH) {
    v = o_prev[b * HH + (k - ED)];
  } else {
    v = h[b * HH + (k - ED - HH)];
  }
  z[idx] = (h16)v;
}

__device__ __forceinline__ float sigmoidf_(float x) { return 1.0f / (1.0f + expf(-x)); }

// gates (64 x 4096) -> h,c in place
__global__ void lstm_elem_k(const float* __restrict__ gates,
                            float* __restrict__ h, float* __restrict__ c) {
  int idx = blockIdx.x * blockDim.x + threadIdx.x;
  if (idx >= BB * HH) return;
  int b = idx >> 10, j = idx & 1023;
  const float* gb = gates + (size_t)b * 4 * HH;
  float ig = sigmoidf_(gb[j]);
  float fg = sigmoidf_(gb[HH + j]);
  float gg = tanhf(gb[2 * HH + j]);
  float og = sigmoidf_(gb[3 * HH + j]);
  float cn = fg * c[idx] + ig * gg;
  c[idx] = cn;
  h[idx] = og * tanhf(cn);
}

// Attention: one block per batch element b (256 threads = 8 waves).
// e[s]=proj[b,s,:].h ; softmax over s ; ctx = sum_s a[s]*enc[b,s,:] ;
// combined_f16[b] = [ctx (2048) | h (1024)]
__global__ __launch_bounds__(256)
void attention_k(const float* __restrict__ proj, const float* __restrict__ enc,
                 const unsigned char* __restrict__ masks,
                 const float* __restrict__ h, h16* __restrict__ combined) {
  const int b = blockIdx.x;
  __shared__ float sh_h[HH];
  __shared__ float sh_e[SRC_];
  const int t = threadIdx.x;
  for (int i = t; i < HH; i += 256) sh_h[i] = h[b * HH + i];
  __syncthreads();

  const int wave = t >> 5, lane = t & 31;
#pragma unroll
  for (int j = 0; j < 8; ++j) {
    int s = wave + 8 * j;
    const float* pr = proj + ((size_t)(b * SRC_ + s)) * HH;
    float p = 0.0f;
    for (int k = lane; k < HH; k += 32) p += pr[k] * sh_h[k];
#pragma unroll
    for (int off = 16; off > 0; off >>= 1) p += __shfl_xor(p, off, 32);
    if (lane == 0) sh_e[s] = masks[b * SRC_ + s] ? -INFINITY : p;
  }
  __syncthreads();

  if (t == 0) {
    float mx = -INFINITY;
    for (int s = 0; s < SRC_; ++s) mx = fmaxf(mx, sh_e[s]);
    float sum = 0.0f;
    for (int s = 0; s < SRC_; ++s) { float e = expf(sh_e[s] - mx); sh_e[s] = e; sum += e; }
    float inv = 1.0f / sum;
    for (int s = 0; s < SRC_; ++s) sh_e[s] *= inv;
  }
  __syncthreads();

  const float* encb = enc + (size_t)b * SRC_ * (2 * HH);
  for (int d = t; d < 2 * HH; d += 256) {
    float acc = 0.0f;
    for (int s = 0; s < SRC_; ++s) acc += sh_e[s] * encb[(size_t)s * (2 * HH) + d];
    combined[(size_t)b * 3 * HH + d] = (h16)acc;
  }
  for (int d = t; d < HH; d += 256)
    combined[(size_t)b * 3 * HH + 2 * HH + d] = (h16)sh_h[d];
}

// log-softmax over each row of 32000 (one block per row)
__global__ __launch_bounds__(256)
void logsoftmax_k(float* __restrict__ X, int ncols) {
  __shared__ float red[256];
  float* row = X + (size_t)blockIdx.x * ncols;
  const int t = threadIdx.x;
  float m = -INFINITY;
  for (int i = t; i < ncols; i += 256) m = fmaxf(m, row[i]);
  red[t] = m; __syncthreads();
  for (int s = 128; s > 0; s >>= 1) { if (t < s) red[t] = fmaxf(red[t], red[t + s]); __syncthreads(); }
  const float rowmax = red[0]; __syncthreads();
  float sum = 0.0f;
  for (int i = t; i < ncols; i += 256) sum += expf(row[i] - rowmax);
  red[t] = sum; __syncthreads();
  for (int s = 128; s > 0; s >>= 1) { if (t < s) red[t] += red[t + s]; __syncthreads(); }
  const float lz = rowmax + logf(red[0]);
  for (int i = t; i < ncols; i += 256) row[i] -= lz;
}

// ---------------------------------------------------------------------------
// Host orchestration
// ---------------------------------------------------------------------------
extern "C" void kernel_launch(void* const* d_in, const int* in_sizes, int n_in,
                              void* d_out, int out_size, void* d_ws, size_t ws_size,
                              hipStream_t stream) {
  const float* enc      = (const float*)d_in[0];   // (64,64,2048)
  const float* h0       = (const float*)d_in[1];   // (64,1024)
  const float* c0       = (const float*)d_in[2];   // (64,1024)
  const float* E        = (const float*)d_in[3];   // (32000,512)
  const float* W_attn   = (const float*)d_in[4];   // (1024,2048)
  const float* W_ih     = (const float*)d_in[5];   // (4096,1536)
  const float* W_hh     = (const float*)d_in[6];   // (4096,1024)
  const float* b_ih     = (const float*)d_in[7];   // (4096,)
  const float* b_hh     = (const float*)d_in[8];   // (4096,)
  const float* W_comb   = (const float*)d_in[9];   // (1024,3072)
  const float* W_vocab  = (const float*)d_in[10];  // (32000,1024)
  const unsigned char* masks = (const unsigned char*)d_in[11]; // (64,64) bool
  const int*   tgt      = (const int*)d_in[12];    // (33,64)
  float* out = (float*)d_out;                      // (32,64,32000)

  // Workspace carve-up
  char* ws = (char*)d_ws;
  size_t off = 0;
  auto alloc = [&](size_t bytes) -> void* {
    void* p = ws + off;
    off = (off + bytes + 255) & ~(size_t)255;
    return p;
  };
  h16*   encA_f16  = (h16*)  alloc((size_t)BB * SRC_ * 2 * HH * 2);     // 16.8 MB
  h16*   Wattn_f16 = (h16*)  alloc((size_t)HH * 2 * HH * 2);            //  4.2 MB
  h16*   Wz_f16    = (h16*)  alloc((size_t)4 * HH * 2560 * 2);          // 21.0 MB
  h16*   Wcomb_f16 = (h16*)  alloc((size_t)HH * 3 * HH * 2);            //  6.3 MB
  h16*   Wvoc_f16  = (h16*)  alloc((size_t)VV * HH * 2);                // 65.5 MB
  float* proj      = (float*)alloc((size_t)BB * SRC_ * HH * 4);         // 16.8 MB
  float* bsum      = (float*)alloc((size_t)4 * HH * 4);
  float* hbuf      = (float*)alloc((size_t)BB * HH * 4);
  float* cbuf      = (float*)alloc((size_t)BB * HH * 4);
  float* obuf      = (float*)alloc((size_t)BB * HH * 4);
  h16*   zbuf      = (h16*)  alloc((size_t)BB * 2560 * 2);
  float* gates     = (float*)alloc((size_t)BB * 4 * HH * 4);
  h16*   comb      = (h16*)  alloc((size_t)BB * 3 * HH * 2);
  h16*   outs_f16  = (h16*)  alloc((size_t)NSTEP * BB * HH * 2);        //  4.2 MB
  (void)ws_size; (void)n_in; (void)in_sizes; (void)out_size;

  const int T256 = 256;
  auto blocks = [](size_t n) { return (unsigned)((n + 255) / 256); };

  // ---- One-time conversions ----
  f32_to_f16_k<<<blocks((size_t)BB * SRC_ * 2 * HH), T256, 0, stream>>>(enc, encA_f16, BB * SRC_ * 2 * HH);
  f32_to_f16_k<<<blocks((size_t)HH * 2 * HH), T256, 0, stream>>>(W_attn, Wattn_f16, HH * 2 * HH);
  f32_to_f16_k<<<blocks((size_t)HH * 3 * HH), T256, 0, stream>>>(W_comb, Wcomb_f16, HH * 3 * HH);
  f32_to_f16_k<<<blocks((size_t)VV * HH), T256, 0, stream>>>(W_vocab, Wvoc_f16, VV * HH);
  build_wz_k<<<blocks((size_t)4 * HH * 2560), T256, 0, stream>>>(W_ih, W_hh, Wz_f16, 4 * HH * 2560);
  bias_sum_k<<<blocks(4 * HH), T256, 0, stream>>>(b_ih, b_hh, bsum, 4 * HH);
  init_state_k<<<blocks(BB * HH), T256, 0, stream>>>(h0, c0, hbuf, cbuf, obuf, BB * HH);

  // ---- proj = enc @ W_attn^T : (4096 x 2048)*(2048 x 1024) ----
  {
    dim3 grid(HH / 64, (BB * SRC_) / 64);
    gemm_f16_wmma<<<grid, 128, 0, stream>>>(encA_f16, Wattn_f16, nullptr, proj,
                                            BB * SRC_, HH, 2 * HH, 0);
  }

  // ---- 32 sequential decode steps ----
  for (int t = 0; t < NSTEP; ++t) {
    build_z_k<<<blocks((size_t)BB * 2560), T256, 0, stream>>>(E, tgt, obuf, hbuf, zbuf, t);

    {   // gates = z @ Wz^T + (b_ih+b_hh) : (64 x 2560)*(2560 x 4096)
      dim3 grid((4 * HH) / 64, BB / 64);
      gemm_f16_wmma<<<grid, 128, 0, stream>>>(zbuf, Wz_f16, bsum, gates,
                                              BB, 4 * HH, 2560, 0);
    }

    lstm_elem_k<<<blocks(BB * HH), T256, 0, stream>>>(gates, hbuf, cbuf);

    attention_k<<<BB, 256, 0, stream>>>(proj, enc, masks, hbuf, comb);

    {   // o_t = tanh(combined @ W_comb^T) : (64 x 3072)*(3072 x 1024)
      dim3 grid(HH / 64, BB / 64);
      gemm_f16_wmma<<<grid, 128, 0, stream>>>(comb, Wcomb_f16, nullptr, obuf,
                                              BB, HH, 3 * HH, 1);
    }

    f32_to_f16_k<<<blocks(BB * HH), T256, 0, stream>>>(obuf, outs_f16 + (size_t)t * BB * HH, BB * HH);
  }

  // ---- logits = outs @ W_vocab^T : (2048 x 1024)*(1024 x 32000) -> d_out ----
  {
    dim3 grid(VV / 64, (NSTEP * BB) / 64);
    gemm_f16_wmma<<<grid, 128, 0, stream>>>(outs_f16, Wvoc_f16, nullptr, out,
                                            NSTEP * BB, VV, HH, 0);
  }

  // ---- in-place log-softmax over V ----
  logsoftmax_k<<<NSTEP * BB, 256, 0, stream>>>(out, VV);
}
